// my_enc_53936199303319
// MI455X (gfx1250) — matrix operand order repaired and checked by reading, not compiled, not run on hardware
//
#include <hip/hip_runtime.h>

#define BATCH  2
#define SEQ    2048
#define HIDSZ  1024
#define NHEAD  16
#define HDIM   64
#define FFNSZ  4096
#define BSROWS (BATCH*SEQ)   // 4096
#define MWORDS (SEQ/32)      // mask words per row = 64

typedef _Float16 v16h __attribute__((ext_vector_type(16)));
typedef _Float16 v8h  __attribute__((ext_vector_type(8)));
typedef float    v8f  __attribute__((ext_vector_type(8)));
typedef unsigned int v4u __attribute__((ext_vector_type(4)));
typedef int      v8i  __attribute__((ext_vector_type(8)));
typedef int      v4i  __attribute__((ext_vector_type(4)));

static __device__ __forceinline__ v16h pack16(v8h lo, v8h hi) {
  v16h r;
#pragma unroll
  for (int i = 0; i < 8; ++i) { r[i] = lo[i]; r[i + 8] = hi[i]; }
  return r;
}

static __device__ __forceinline__ v8f wmma16(v16h a, v16h b, v8f c) {
  // D = A(16x32 f16) x B(32x16 f16) + C(16x16 f32)
  return __builtin_amdgcn_wmma_f32_16x16x32_f16(false, a, false, b, (short)0, c,
                                                false, false);
}

// ---------------------------------------------------------------------------
// TDM: 2D tile global->LDS via Tensor Data Mover (CDNA5 ISA ch.8 descriptor)
// dims in elements (data_size = 2 bytes / f16)
// ---------------------------------------------------------------------------
static __device__ __forceinline__ void tdm_load_2d_f16(
    unsigned int ldsOff, const void* gptr, unsigned int tileD0,
    unsigned int tileD1, unsigned int tenD0, unsigned int tenD1,
    unsigned int stride0) {
  unsigned long long ga = (unsigned long long)(uintptr_t)gptr;
  v4u g0;
  g0[0] = 1u;                                   // count = 1 valid descriptor
  g0[1] = ldsOff;                               // lds_addr (bytes)
  g0[2] = (unsigned int)ga;                     // global_addr[31:0]
  g0[3] = (unsigned int)((ga >> 32) & 0x1FFFFFFu) | (2u << 30);  // addr hi | type=2
  v8i g1;
  g1[0] = (int)(1u << 16);                      // data_size = 1 -> 2 bytes
  g1[1] = (int)(tenD0 << 16);                   // tensor_dim0[15:0]
  g1[2] = (int)((tenD0 >> 16) | (tenD1 << 16)); // tensor_dim0 hi | tensor_dim1 lo
  g1[3] = (int)((tenD1 >> 16) | (tileD0 << 16));// tensor_dim1 hi | tile_dim0
  g1[4] = (int)(tileD1 & 0xFFFFu);              // tile_dim1 (tile_dim2 = 0)
  g1[5] = (int)stride0;                         // tensor_dim0_stride[31:0]
  g1[6] = 0;
  g1[7] = 0;
  v4i gz = {0, 0, 0, 0};
#if __clang_major__ >= 23
  v8i gz8 = {0, 0, 0, 0, 0, 0, 0, 0};
  __builtin_amdgcn_tensor_load_to_lds(g0, g1, gz, gz, gz8, 0);
#else
  __builtin_amdgcn_tensor_load_to_lds(g0, g1, gz, gz, 0);
#endif
}

static __device__ __forceinline__ unsigned int lds_offset(const void* p) {
  // generic LDS pointer: low 32 bits are the LDS byte offset
  return (unsigned int)(uintptr_t)p;
}

// ---------------------------------------------------------------------------
// f32 -> f16 weight conversion
// ---------------------------------------------------------------------------
__global__ void cvt_f32_to_f16(const float* __restrict__ x,
                               _Float16* __restrict__ y, int n) {
  for (int i = blockIdx.x * blockDim.x + threadIdx.x; i < n;
       i += gridDim.x * blockDim.x)
    y[i] = (_Float16)x[i];
}

// ---------------------------------------------------------------------------
// Pack bool mask bytes into bitmask words (32 keys per u32)
// ---------------------------------------------------------------------------
__global__ void pack_mask(const unsigned char* __restrict__ m,
                          unsigned int* __restrict__ bits, int nwords) {
  for (int i = blockIdx.x * blockDim.x + threadIdx.x; i < nwords;
       i += gridDim.x * blockDim.x) {
    const uint4* p4 = (const uint4*)(m + (size_t)i * 32);
    uint4 a = p4[0], b = p4[1];
    unsigned int dw[8] = {a.x, a.y, a.z, a.w, b.x, b.y, b.z, b.w};
    unsigned int w = 0;
#pragma unroll
    for (int d = 0; d < 8; ++d)
#pragma unroll
      for (int k = 0; k < 4; ++k)
        w |= (((dw[d] >> (8 * k)) & 0xFFu) ? 1u : 0u) << (d * 4 + k);
    bits[i] = w;
  }
}

// ---------------------------------------------------------------------------
// WMMA GEMM with TDM-staged weights:
//   out[M,N] = act( A[M,K] * W[N,K]^T + bias[N] ) (+ resid)
//   block: 256 threads = 8 waves as 2(M) x 4(N); wave tile 32x64;
//   block tile 64 x 256; K step 32.
//   Wave0 TDM-loads the next 256x32 W tile into double-buffered LDS while all
//   waves run WMMAs on the current buffer (s_wait_tensorcnt + barrier).
//   A fragments are loaded directly from global (f32 converted on load).
// ---------------------------------------------------------------------------
template <bool A32P, bool RELU, bool VTRANS>
__global__ __launch_bounds__(256) void gemm_wmma(
    const float* __restrict__ A32, const _Float16* __restrict__ A16,
    const _Float16* __restrict__ W, const float* __restrict__ bias,
    float* __restrict__ outF32, _Float16* __restrict__ outF16,
    const float* __restrict__ resid, int M, int N, int K) {
  const int lane = threadIdx.x & 31;
  const int wave = threadIdx.x >> 5;
  const int lr = lane & 15;   // row/col within fragment
  const int hh = lane >> 4;   // half-wave selector
  const int mBase = blockIdx.y * 64 + (wave >> 2) * 32;
  const int nBase = blockIdx.x * 256 + (wave & 3) * 64;
  const int nLoc = (wave & 3) * 64;  // wave's N offset inside the LDS tile

  __shared__ _Float16 Wl[2][256][32];  // double-buffered weight tile (32 KB)

  const v8f vzero = {0.f, 0.f, 0.f, 0.f, 0.f, 0.f, 0.f, 0.f};
  v8f acc[2][4];
#pragma unroll
  for (int mi = 0; mi < 2; ++mi)
#pragma unroll
    for (int ni = 0; ni < 4; ++ni) acc[mi][ni] = vzero;

  const _Float16* wBase = W + (size_t)(blockIdx.x * 256) * K;

  // prologue: issue TDM for the first weight tile
  if (wave == 0)
    tdm_load_2d_f16(lds_offset(&Wl[0][0][0]), wBase, 32, 256, 32, 256, K);

  for (int k = 0; k < K; k += 32) {
    const int buf = (k >> 5) & 1;
    if (wave == 0) __builtin_amdgcn_s_wait_tensorcnt(0);
    __syncthreads();
    if (wave == 0 && k + 32 < K)
      tdm_load_2d_f16(lds_offset(&Wl[buf ^ 1][0][0]), wBase + k + 32, 32, 256,
                      32, 256, K);

    v16h af[2];
#pragma unroll
    for (int mi = 0; mi < 2; ++mi) {
      const int row = mBase + mi * 16 + lr;
      if (A32P) {
        const float* ap = A32 + (size_t)row * K + k + hh * 8;
        float4 a0 = *(const float4*)(ap + 0);
        float4 a1 = *(const float4*)(ap + 4);
        float4 a2 = *(const float4*)(ap + 16);
        float4 a3 = *(const float4*)(ap + 20);
        v16h t;
        t[0] = (_Float16)a0.x;  t[1] = (_Float16)a0.y;
        t[2] = (_Float16)a0.z;  t[3] = (_Float16)a0.w;
        t[4] = (_Float16)a1.x;  t[5] = (_Float16)a1.y;
        t[6] = (_Float16)a1.z;  t[7] = (_Float16)a1.w;
        t[8] = (_Float16)a2.x;  t[9] = (_Float16)a2.y;
        t[10] = (_Float16)a2.z; t[11] = (_Float16)a2.w;
        t[12] = (_Float16)a3.x; t[13] = (_Float16)a3.y;
        t[14] = (_Float16)a3.z; t[15] = (_Float16)a3.w;
        af[mi] = t;
      } else {
        const _Float16* ap = A16 + (size_t)row * K + k + hh * 8;
        af[mi] = pack16(*(const v8h*)ap, *(const v8h*)(ap + 16));
      }
    }
    v16h bf[4];
#pragma unroll
    for (int ni = 0; ni < 4; ++ni) {
      const _Float16* bp = &Wl[buf][nLoc + ni * 16 + lr][hh * 8];
      bf[ni] = pack16(*(const v8h*)bp, *(const v8h*)(bp + 16));
    }
#pragma unroll
    for (int mi = 0; mi < 2; ++mi)
#pragma unroll
      for (int ni = 0; ni < 4; ++ni)
        acc[mi][ni] = wmma16(af[mi], bf[ni], acc[mi][ni]);
  }

#pragma unroll
  for (int mi = 0; mi < 2; ++mi)
#pragma unroll
    for (int ni = 0; ni < 4; ++ni) {
      const int col = nBase + ni * 16 + lr;
      const float bv = bias[col];
#pragma unroll
      for (int r = 0; r < 8; ++r) {
        const int row = mBase + mi * 16 + hh * 8 + r;
        float v = acc[mi][ni][r] + bv;
        if (RELU) v = fmaxf(v, 0.f);
        if (resid) v += resid[(size_t)row * N + col];
        if (outF32) outF32[(size_t)row * N + col] = v;
        if (outF16) {
          if (VTRANS) {
            const int s = row & (SEQ - 1);
            const int b = row >> 11;
            const int h = col >> 6;
            const int dl = col & (HDIM - 1);
            outF16[((size_t)((b * NHEAD + h) * HDIM + dl)) * SEQ + s] =
                (_Float16)v;
          } else {
            outF16[(size_t)row * N + col] = (_Float16)v;
          }
        }
      }
    }
}

// ---------------------------------------------------------------------------
// Flash attention with TDM double-buffered K/V tiles.
// Block = 4 waves x 16 query rows = 64 q rows of one (b,h).
// Per 32-key tile: wave0 TDM-loads next K (32x64) and V^T (64x32) tiles to
// LDS; all waves do QK^T (4 WMMA), online softmax (packed bitmask), P
// re-layout via wave-private LDS, PV (4 WMMA).
// ---------------------------------------------------------------------------
__global__ __launch_bounds__(128) void flash_attn(
    const _Float16* __restrict__ Qh, const _Float16* __restrict__ Kh,
    const _Float16* __restrict__ Vt, const unsigned int* __restrict__ Mbits,
    _Float16* __restrict__ ctx) {
  const int qt = blockIdx.x;
  const int bh = blockIdx.y;
  const int b = bh / NHEAD, h = bh % NHEAD;
  const int lane = threadIdx.x & 31;
  const int wave = threadIdx.x >> 5;
  const int lr = lane & 15;
  const int hh = lane >> 4;
  const int qw = qt * 64 + wave * 16;  // first query row of this wave
  const float scale = 0.125f;          // 1/sqrt(64)

  __shared__ _Float16 Kt[2][32][64];   // key tiles  (rows=key, cols=d)
  __shared__ _Float16 Vl[2][64][32];   // V^T tiles  (rows=d,   cols=key)
  __shared__ v8h Pl[4][16][4];         // wave-private P staging

  // Q A-fragments (16 x 64 = two K=32 chunks), resident for whole loop
  v16h qa[2];
  {
    const _Float16* qp = Qh + ((size_t)(b * SEQ + qw + lr)) * HIDSZ + h * HDIM;
#pragma unroll
    for (int kb = 0; kb < 2; ++kb) {
      const _Float16* p = qp + kb * 32 + hh * 8;
      qa[kb] = pack16(*(const v8h*)p, *(const v8h*)(p + 16));
    }
  }

  // packed mask: this lane serves row (hh*8 + (lr&7)) of the wave's 16 rows
  const unsigned int* mrowp =
      Mbits + (size_t)(b * SEQ + qw + hh * 8 + (lr & 7)) * MWORDS;

  const v8f vzero = {0.f, 0.f, 0.f, 0.f, 0.f, 0.f, 0.f, 0.f};
  v8f acc[4] = {vzero, vzero, vzero, vzero};
  float mx[8], lsum[8];
#pragma unroll
  for (int r = 0; r < 8; ++r) { mx[r] = -1e30f; lsum[r] = 0.f; }

  const _Float16* kBase = Kh + (size_t)(b * SEQ) * HIDSZ + h * HDIM;
  const _Float16* vBase = Vt + (size_t)((b * NHEAD + h) * HDIM) * SEQ;

  // prologue: issue TDM for tile 0
  if (wave == 0) {
    tdm_load_2d_f16(lds_offset(&Kt[0][0][0]), kBase, 64, 32, 64, 32, HIDSZ);
    tdm_load_2d_f16(lds_offset(&Vl[0][0][0]), vBase, 32, 64, 32, 64, SEQ);
  }

  const int NT = SEQ / 32;
  for (int t = 0; t < NT; ++t) {
    const int buf = t & 1;
    if (wave == 0) __builtin_amdgcn_s_wait_tensorcnt(0);
    __syncthreads();
    if (wave == 0 && t + 1 < NT) {
      const int nb = buf ^ 1;
      const int j1 = (t + 1) * 32;
      tdm_load_2d_f16(lds_offset(&Kt[nb][0][0]),
                      kBase + (size_t)j1 * HIDSZ, 64, 32, 64, 32, HIDSZ);
      tdm_load_2d_f16(lds_offset(&Vl[nb][0][0]), vBase + j1, 32, 64, 32, 64,
                      SEQ);
    }

    // K B-fragments from LDS: B[k=d, n=key]
    v16h kf[2][2];
#pragma unroll
    for (int jt = 0; jt < 2; ++jt)
#pragma unroll
      for (int kb = 0; kb < 2; ++kb) {
        const _Float16* p = &Kt[buf][jt * 16 + lr][kb * 32 + hh * 8];
        kf[jt][kb] = pack16(*(const v8h*)p, *(const v8h*)(p + 16));
      }
    v8f s0 = vzero, s1 = vzero;
    s0 = wmma16(qa[0], kf[0][0], s0);
    s0 = wmma16(qa[1], kf[0][1], s0);
    s1 = wmma16(qa[0], kf[1][0], s1);
    s1 = wmma16(qa[1], kf[1][1], s1);

    // one mask word per lane per tile; rows broadcast via shuffles
    const unsigned int myword = mrowp[t];

#pragma unroll
    for (int r = 0; r < 8; ++r) {
      const unsigned int wr =
          (unsigned int)__shfl((int)myword, hh * 16 + r, 32);
      float va = ((wr >> lr) & 1u) ? -1e9f : s0[r] * scale;
      float vb = ((wr >> (16 + lr)) & 1u) ? -1e9f : s1[r] * scale;
      float rm = fmaxf(va, vb);
#pragma unroll
      for (int o = 8; o >= 1; o >>= 1) rm = fmaxf(rm, __shfl_xor(rm, o, 32));
      const float mnew = fmaxf(mx[r], rm);
      const float al = __expf(mx[r] - mnew);
      const float pa = __expf(va - mnew);
      const float pb = __expf(vb - mnew);
      float rs = pa + pb;
#pragma unroll
      for (int o = 8; o >= 1; o >>= 1) rs += __shfl_xor(rs, o, 32);
      lsum[r] = lsum[r] * al + rs;
      mx[r] = mnew;
#pragma unroll
      for (int d = 0; d < 4; ++d) acc[d][r] *= al;
      Pl[wave][hh * 8 + r][lr >> 3][lr & 7] = (_Float16)pa;
      Pl[wave][hh * 8 + r][2 + (lr >> 3)][lr & 7] = (_Float16)pb;
    }
    // re-read P as 16x32 A fragment (wave-private; DS ops are in order)
    v16h pf = pack16(Pl[wave][lr][hh], Pl[wave][lr][2 + hh]);

    // V B-fragments from LDS V^T tile: B[k=key, n=d], contiguous along keys
#pragma unroll
    for (int d = 0; d < 4; ++d) {
      const _Float16* vp = &Vl[buf][d * 16 + lr][hh * 8];
      v16h vf = pack16(*(const v8h*)vp, *(const v8h*)(vp + 16));
      acc[d] = wmma16(pf, vf, acc[d]);
    }
  }

  // normalize and store context as f16 [B*S, HID]
#pragma unroll
  for (int r = 0; r < 8; ++r) {
    const float inv = 1.f / lsum[r];
    const size_t row = (size_t)(b * SEQ + qw + hh * 8 + r);
#pragma unroll
    for (int d = 0; d < 4; ++d)
      ctx[row * HIDSZ + h * HDIM + d * 16 + lr] = (_Float16)(acc[d][r] * inv);
  }
}

// ---------------------------------------------------------------------------
extern "C" void kernel_launch(void* const* d_in, const int* in_sizes, int n_in,
                              void* d_out, int out_size, void* d_ws,
                              size_t ws_size, hipStream_t stream) {
  (void)in_sizes; (void)n_in; (void)out_size; (void)ws_size;
  const float* srcQ = (const float*)d_in[0];
  const float* srcK = (const float*)d_in[1];
  const float* srcV = (const float*)d_in[2];
  const unsigned char* amask = (const unsigned char*)d_in[3];
  const float* wq = (const float*)d_in[4];  const float* bq = (const float*)d_in[5];
  const float* wk = (const float*)d_in[6];  const float* bk = (const float*)d_in[7];
  const float* wv = (const float*)d_in[8];  const float* bv = (const float*)d_in[9];
  const float* wo = (const float*)d_in[10]; const float* bo = (const float*)d_in[11];
  const float* w1 = (const float*)d_in[12]; const float* b1 = (const float*)d_in[13];
  const float* w2 = (const float*)d_in[14]; const float* b2 = (const float*)d_in[15];
  float* out = (float*)d_out;

  char* base = (char*)d_ws;
  size_t off = 0;
  auto take = [&](size_t bytes) -> char* {
    char* r = base + off;
    off += (bytes + 255) & ~(size_t)255;
    return r;
  };
  _Float16* wqh   = (_Float16*)take((size_t)HIDSZ * HIDSZ * 2);
  _Float16* wkh   = (_Float16*)take((size_t)HIDSZ * HIDSZ * 2);
  _Float16* wvh   = (_Float16*)take((size_t)HIDSZ * HIDSZ * 2);
  _Float16* woh   = (_Float16*)take((size_t)HIDSZ * HIDSZ * 2);
  _Float16* w1h   = (_Float16*)take((size_t)FFNSZ * HIDSZ * 2);
  _Float16* w2h   = (_Float16*)take((size_t)FFNSZ * HIDSZ * 2);
  _Float16* Qh    = (_Float16*)take((size_t)BSROWS * HIDSZ * 2);
  _Float16* Kh    = (_Float16*)take((size_t)BSROWS * HIDSZ * 2);
  _Float16* Vt    = (_Float16*)take((size_t)BSROWS * HIDSZ * 2);
  _Float16* ctx   = (_Float16*)take((size_t)BSROWS * HIDSZ * 2);
  _Float16* attnh = (_Float16*)take((size_t)BSROWS * HIDSZ * 2);
  float*    attnf = (float*)take((size_t)BSROWS * HIDSZ * 4);
  _Float16* h1    = (_Float16*)take((size_t)BSROWS * FFNSZ * 2);
  unsigned int* mbits =
      (unsigned int*)take((size_t)BATCH * SEQ * MWORDS * 4);

  // weight conversion f32 -> f16
  cvt_f32_to_f16<<<1024, 256, 0, stream>>>(wq, wqh, HIDSZ * HIDSZ);
  cvt_f32_to_f16<<<1024, 256, 0, stream>>>(wk, wkh, HIDSZ * HIDSZ);
  cvt_f32_to_f16<<<1024, 256, 0, stream>>>(wv, wvh, HIDSZ * HIDSZ);
  cvt_f32_to_f16<<<1024, 256, 0, stream>>>(wo, woh, HIDSZ * HIDSZ);
  cvt_f32_to_f16<<<1024, 256, 0, stream>>>(w1, w1h, FFNSZ * HIDSZ);
  cvt_f32_to_f16<<<1024, 256, 0, stream>>>(w2, w2h, FFNSZ * HIDSZ);
  // mask bit-packing
  pack_mask<<<512, 256, 0, stream>>>(amask, mbits, BATCH * SEQ * MWORDS);

  const dim3 blk(256);
  const dim3 gHID(HIDSZ / 256, BSROWS / 64);
  const dim3 gFFN(FFNSZ / 256, BSROWS / 64);

  // Q/K/V projections (A is f32 source, converted on load)
  gemm_wmma<true, false, false><<<gHID, blk, 0, stream>>>(
      srcQ, nullptr, wqh, bq, nullptr, Qh, nullptr, BSROWS, HIDSZ, HIDSZ);
  gemm_wmma<true, false, false><<<gHID, blk, 0, stream>>>(
      srcK, nullptr, wkh, bk, nullptr, Kh, nullptr, BSROWS, HIDSZ, HIDSZ);
  gemm_wmma<true, false, true><<<gHID, blk, 0, stream>>>(
      srcV, nullptr, wvh, bv, nullptr, Vt, nullptr, BSROWS, HIDSZ, HIDSZ);

  // attention
  flash_attn<<<dim3(SEQ / 64, BATCH * NHEAD), 128, 0, stream>>>(Qh, Kh, Vt,
                                                                mbits, ctx);

  // output projection (residual source, kept in both f32 and f16)
  gemm_wmma<false, false, false><<<gHID, blk, 0, stream>>>(
      nullptr, ctx, woh, bo, attnf, attnh, nullptr, BSROWS, HIDSZ, HIDSZ);

  // FFN
  gemm_wmma<false, true, false><<<gFFN, blk, 0, stream>>>(
      nullptr, attnh, w1h, b1, nullptr, h1, nullptr, BSROWS, FFNSZ, HIDSZ);
  gemm_wmma<false, false, false><<<gHID, blk, 0, stream>>>(
      nullptr, h1, w2h, b2, out, nullptr, attnf, BSROWS, HIDSZ, FFNSZ);
}
